// ReLiKReader_66348654788737
// MI455X (gfx1250) — matrix operand order repaired and checked by reading, not compiled, not run on hardware
//
#include <hip/hip_runtime.h>
#include <hip/hip_bf16.h>
#include <math.h>

// ---------------------------------------------------------------------------
// Model constants (ReLiK reader mini-BERT layer)
// ---------------------------------------------------------------------------
#define BB   4
#define SS   1024
#define HH   768
#define NHH  12
#define DHD  64          // head dim
#define DFFC 3072
#define MTOK 4096        // B*S

typedef __attribute__((ext_vector_type(16))) __bf16 v16bf;
typedef __attribute__((ext_vector_type(8)))  float  v8f;

union FragU {
    v16bf v;
    __bf16 h[16];
    uint4  q[2];
};

// ---------------------------------------------------------------------------
// WMMA helper  (wave32, D = A(16x32 bf16) * B(32x16 bf16) + C(16x16 f32))
// ---------------------------------------------------------------------------
__device__ __forceinline__ v8f wmma_bf16(v16bf a, v16bf b, v8f c) {
    return __builtin_amdgcn_wmma_f32_16x16x32_bf16(
        /*neg_a=*/false, a, /*neg_b=*/false, b,
        /*c_mod=*/(short)0, c, /*reuse_a=*/false, /*reuse_b=*/false);
}

// A fragment from a row-major bf16 matrix in GLOBAL memory.
__device__ __forceinline__ v16bf load_a_frag(const __bf16* A, int lda,
                                             int m0, int k0, int lane) {
    int m = m0 + (lane & 15);
    int g = lane >> 4;
    const __bf16* p = A + (size_t)m * lda + k0 + 8 * g;
    FragU f;
    f.q[0] = *(const uint4*)(p);
    f.q[1] = *(const uint4*)(p + 16);
    return f.v;
}

// A fragment from an LDS tile laid out as [rows][32 bf16] (64B per row).
__device__ __forceinline__ v16bf load_a_frag_tile(const __bf16* T,
                                                  int m0, int lane) {
    int m = m0 + (lane & 15);
    int g = lane >> 4;
    const char* p = (const char*)(T + m * 32) + 16 * g;
    FragU f;
    f.q[0] = *(const uint4*)(p);
    f.q[1] = *(const uint4*)(p + 32);
    return f.v;
}

// B fragment: 32x16 tile from a K-contiguous ("transposed") bf16 matrix.
__device__ __forceinline__ v16bf load_b_frag(const __bf16* Bt, int ldk,
                                             int n0, int k0, int lane) {
    int n = n0 + (lane & 15);
    int g = lane >> 4;
    const __bf16* p = Bt + (size_t)n * ldk + k0 + 8 * g;
    FragU f;
    f.q[0] = *(const uint4*)(p);
    f.q[1] = *(const uint4*)(p + 16);
    return f.v;
}

// A fragment sourced from f32 scores in LDS (attention probabilities).
__device__ __forceinline__ v16bf load_a_frag_lds_f32(const float* S, int ld,
                                                     int k0, int lane) {
    int m = lane & 15;
    int g = lane >> 4;
    const float* p = S + m * ld + k0 + 8 * g;
    FragU f;
#pragma unroll
    for (int e = 0; e < 8; ++e) f.h[e]     = (__bf16)p[e];
#pragma unroll
    for (int e = 0; e < 8; ++e) f.h[8 + e] = (__bf16)p[16 + e];
    return f.v;
}

// ---------------------------------------------------------------------------
// CDNA5 async global->LDS (ASYNCcnt path).  One B128 per lane.
// ---------------------------------------------------------------------------
__device__ __forceinline__ void async_load_b128(unsigned lds_off,
                                                const void* gptr) {
    asm volatile("global_load_async_to_lds_b128 %0, %1, off"
                 :
                 : "v"(lds_off), "v"((unsigned long long)(uintptr_t)gptr)
                 : "memory");
}
__device__ __forceinline__ void wait_async_le1() {
    asm volatile("s_wait_asynccnt 0x1" ::: "memory");
}
__device__ __forceinline__ void wait_async_0() {
    asm volatile("s_wait_asynccnt 0x0" ::: "memory");
}

// ---------------------------------------------------------------------------
// Small helpers
// ---------------------------------------------------------------------------
__device__ __forceinline__ float gelu_f(float x) {
    float x3 = x * x * x;
    return 0.5f * x * (1.0f + tanhf(0.7978845608028654f * (x + 0.044715f * x3)));
}

__device__ __forceinline__ float block_reduce_sum(float v) {
    __shared__ float tmp[8];
#pragma unroll
    for (int o = 16; o > 0; o >>= 1) v += __shfl_xor(v, o);
    int lane = threadIdx.x & 31;
    int w    = threadIdx.x >> 5;
    if (lane == 0) tmp[w] = v;
    __syncthreads();
    float r = 0.f;
    if (threadIdx.x < 8) r = tmp[threadIdx.x];
    if (w == 0) {
#pragma unroll
        for (int o = 4; o > 0; o >>= 1) r += __shfl_xor(r, o);
        if (lane == 0) tmp[0] = r;
    }
    __syncthreads();
    float out = tmp[0];
    __syncthreads();
    return out;
}

// ---------------------------------------------------------------------------
// Weight convert+transpose: W f32 [K,N] row-major -> Wt bf16 [N,K]
// ---------------------------------------------------------------------------
__global__ void wt_convert_kernel(const float* __restrict__ W,
                                  __bf16* __restrict__ Wt, int K, int N) {
    size_t total = (size_t)K * N;
    for (size_t i = blockIdx.x * (size_t)blockDim.x + threadIdx.x; i < total;
         i += (size_t)gridDim.x * blockDim.x) {
        size_t k = i / N, n = i % N;
        Wt[n * (size_t)K + k] = (__bf16)W[i];
    }
}

// ---------------------------------------------------------------------------
// Embedding gather + LayerNorm -> f32 and bf16 activations
// ---------------------------------------------------------------------------
__global__ void embed_ln_kernel(const int* __restrict__ ids,
                                const float* __restrict__ embed,
                                const float* __restrict__ g,
                                const float* __restrict__ b,
                                float* __restrict__ y, __bf16* __restrict__ ybf) {
    int t = blockIdx.x;
    const float* row = embed + (size_t)ids[t] * HH;
    float s = 0.f;
    for (int j = threadIdx.x; j < HH; j += blockDim.x) s += row[j];
    float mean = block_reduce_sum(s) * (1.0f / HH);
    float v = 0.f;
    for (int j = threadIdx.x; j < HH; j += blockDim.x) {
        float d = row[j] - mean; v += d * d;
    }
    float var  = block_reduce_sum(v) * (1.0f / HH);
    float rinv = rsqrtf(var + 1e-7f);
    for (int j = threadIdx.x; j < HH; j += blockDim.x) {
        float val = (row[j] - mean) * rinv * g[j] + b[j];
        y  [(size_t)t * HH + j] = val;
        ybf[(size_t)t * HH + j] = (__bf16)val;
    }
}

// ---------------------------------------------------------------------------
// LayerNorm over rows of x -> f32 (+ optional bf16 copy)
// ---------------------------------------------------------------------------
__global__ void ln_kernel(const float* __restrict__ x,
                          const float* __restrict__ g,
                          const float* __restrict__ b,
                          float* __restrict__ y, __bf16* __restrict__ ybf) {
    int t = blockIdx.x;
    const float* row = x + (size_t)t * HH;
    float s = 0.f;
    for (int j = threadIdx.x; j < HH; j += blockDim.x) s += row[j];
    float mean = block_reduce_sum(s) * (1.0f / HH);
    float v = 0.f;
    for (int j = threadIdx.x; j < HH; j += blockDim.x) {
        float d = row[j] - mean; v += d * d;
    }
    float var  = block_reduce_sum(v) * (1.0f / HH);
    float rinv = rsqrtf(var + 1e-7f);
    for (int j = threadIdx.x; j < HH; j += blockDim.x) {
        float val = (row[j] - mean) * rinv * g[j] + b[j];
        y[(size_t)t * HH + j] = val;
        if (ybf) ybf[(size_t)t * HH + j] = (__bf16)val;
    }
}

// ---------------------------------------------------------------------------
// WMMA GEMM:  C[M,N] = act(A[M,K]_bf16 @ Wt[N,K]^T + bias) (+res)
// 256 threads = 8 waves (2 wave-rows x 4 wave-cols).
// Wave tile 32x64 (8 accumulators); block tile 64x256.
// A block-tile (64 rows x 32 K = 4KB) DMA'd to LDS with
// global_load_async_to_lds_b128, double-buffered; B frags stream from the
// L2-resident transposed weights.  __launch_bounds__(256,1): full VGPR file,
// zero spills — latency hidden by the async pipeline, not occupancy.
// ---------------------------------------------------------------------------
__device__ __forceinline__ void gemm_step(const __bf16* tile,
                                          const __bf16* __restrict__ Wt,
                                          int K, int n0, int k0,
                                          int wrow, int lane,
                                          v8f acc[2][4]) {
    // batch the 4 global B loads first, then the 2 LDS A loads,
    // then burst 8 WMMAs (waits retire progressively)
    v16bf b0 = load_b_frag(Wt, K, n0 + 0,  k0, lane);
    v16bf b1 = load_b_frag(Wt, K, n0 + 16, k0, lane);
    v16bf b2 = load_b_frag(Wt, K, n0 + 32, k0, lane);
    v16bf b3 = load_b_frag(Wt, K, n0 + 48, k0, lane);
    v16bf a0 = load_a_frag_tile(tile, wrow * 32,      lane);
    v16bf a1 = load_a_frag_tile(tile, wrow * 32 + 16, lane);
    acc[0][0] = wmma_bf16(a0, b0, acc[0][0]);
    acc[0][1] = wmma_bf16(a0, b1, acc[0][1]);
    acc[0][2] = wmma_bf16(a0, b2, acc[0][2]);
    acc[0][3] = wmma_bf16(a0, b3, acc[0][3]);
    acc[1][0] = wmma_bf16(a1, b0, acc[1][0]);
    acc[1][1] = wmma_bf16(a1, b1, acc[1][1]);
    acc[1][2] = wmma_bf16(a1, b2, acc[1][2]);
    acc[1][3] = wmma_bf16(a1, b3, acc[1][3]);
}

__global__ void __launch_bounds__(256, 1)
gemm_wmma_kernel(const __bf16* __restrict__ A,
                 const __bf16* __restrict__ Wt,
                 const float* __restrict__ bias,
                 int M, int N, int K,
                 float* __restrict__ Cf32,
                 __bf16* __restrict__ Cbf16,
                 const float* __restrict__ res,
                 __bf16* __restrict__ vT,
                 int do_gelu) {
    __shared__ __align__(16) __bf16 atile[2][64 * 32];   // 2 x 4KB

    int tid  = threadIdx.x;
    int lane = tid & 31;
    int wave = tid >> 5;
    int wrow = wave >> 2;                 // 0..1
    int wcol = wave & 3;                  // 0..3
    int row0 = blockIdx.y * 64;           // block M origin
    int n0   = blockIdx.x * 256 + wcol * 64;

    // cooperative async copy: thread t moves 16B: row t>>2, 16B-chunk t&3
    int arow = tid >> 2;
    int achk = tid & 3;
    const __bf16* aptr = A + (size_t)(row0 + arow) * K + achk * 8;
    unsigned ldsdst[2];
    ldsdst[0] = (unsigned)(uintptr_t)&atile[0][0] + arow * 64 + achk * 16;
    ldsdst[1] = (unsigned)(uintptr_t)&atile[1][0] + arow * 64 + achk * 16;

    v8f acc[2][4] = {{{}, {}, {}, {}}, {{}, {}, {}, {}}};

    const int nsteps = K / 32;
    async_load_b128(ldsdst[0], aptr);                 // prologue: k-step 0

    // steady state: branch-free issue(next) -> wait(cur) -> compute(cur)
    for (int s = 0; s < nsteps - 1; ++s) {
        async_load_b128(ldsdst[(s + 1) & 1], aptr + (size_t)(s + 1) * 32);
        wait_async_le1();                             // current buffer landed
        __syncthreads();                              // all waves' DMA visible
        gemm_step(&atile[s & 1][0], Wt, K, n0, s * 32, wrow, lane, acc);
        __syncthreads();          // everyone done reading before next overwrite
    }
    // peeled last step
    wait_async_0();
    __syncthreads();
    gemm_step(&atile[(nsteps - 1) & 1][0], Wt, K, n0, (nsteps - 1) * 32,
              wrow, lane, acc);

    int g  = lane >> 4;
    int nn = lane & 15;
#pragma unroll
    for (int i = 0; i < 2; ++i) {
        int m0 = row0 + wrow * 32 + i * 16;
#pragma unroll
        for (int j = 0; j < 4; ++j) {
            int n = n0 + j * 16 + nn;
            float bv = bias ? bias[n] : 0.f;
#pragma unroll
            for (int r = 0; r < 8; ++r) {
                int mrow = m0 + r + 8 * g;
                float val = acc[i][j][r] + bv;
                if (do_gelu) val = gelu_f(val);
                if (res)   val += res[(size_t)mrow * N + n];
                if (Cf32)  Cf32 [(size_t)mrow * N + n] = val;
                if (Cbf16) Cbf16[(size_t)mrow * N + n] = (__bf16)val;
                if (vT) {  // [b, h, d, s] head-transposed layout (for V)
                    int b = mrow >> 10, sx = mrow & 1023;
                    int h = n >> 6,    d  = n & 63;
                    vT[(((size_t)(b * NHH + h)) * DHD + d) * SS + sx] =
                        (__bf16)val;
                }
            }
        }
    }
}

// ---------------------------------------------------------------------------
// Attention: one wave per (b, head, 16-row q tile).
//   pass1: scores(16x1024) = (Q 16x64) @ K^T / 8 (+mask) -> LDS f32
//   pass2: two-lane-per-row softmax (shfl across half-wave pairs)
//   pass3: ctx(16x64) = P @ V via WMMA, A-frags from LDS
// Dynamic LDS: 16*1024 f32 = 64KB (WGP has 320KB).
// ---------------------------------------------------------------------------
__global__ void __launch_bounds__(32, 1)
attention_kernel(const __bf16* __restrict__ q,
                 const __bf16* __restrict__ k,
                 const __bf16* __restrict__ vT,
                 const int* __restrict__ amask,
                 __bf16* __restrict__ ctx) {
    extern __shared__ float sS[];            // [16][1024]
    int lane = threadIdx.x;
    int qt = blockIdx.x, h = blockIdx.y, b = blockIdx.z;
    int g  = lane >> 4;
    int nn = lane & 15;
    int tok0 = b * SS + qt * 16;

    v16bf qa0 = load_a_frag(q, HH, tok0, h * DHD,      lane);
    v16bf qa1 = load_a_frag(q, HH, tok0, h * DHD + 32, lane);

    const __bf16* kb = k + (size_t)b * SS * HH;
    for (int kt = 0; kt < SS / 16; ++kt) {
        v16bf kb0 = load_b_frag(kb, HH, kt * 16, h * DHD,      lane);
        v16bf kb1 = load_b_frag(kb, HH, kt * 16, h * DHD + 32, lane);
        v8f acc = {};
        acc = wmma_bf16(qa0, kb0, acc);
        acc = wmma_bf16(qa1, kb1, acc);
        int key  = kt * 16 + nn;
        float mv = (amask[b * SS + key] == 0) ? -1.0e9f : 0.f;
#pragma unroll
        for (int r = 0; r < 8; ++r) {
            int mrow = r + 8 * g;
            sS[mrow * SS + key] = acc[r] * 0.125f + mv;   // 1/sqrt(64)
        }
    }
    __syncthreads();

    int row = lane & 15;
    float* srow = sS + row * SS + (lane >> 4) * 512;
    float mx = -3.0e38f;
    for (int j = 0; j < 512; ++j) mx = fmaxf(mx, srow[j]);
    mx = fmaxf(mx, __shfl_xor(mx, 16));
    float sum = 0.f;
    for (int j = 0; j < 512; ++j) {
        float e = __expf(srow[j] - mx);
        srow[j] = e;
        sum += e;
    }
    sum += __shfl_xor(sum, 16);
    float inv = 1.0f / sum;                  // lanes 0..15 hold rows 0..15
    __syncthreads();

    v8f cacc[4] = {{}, {}, {}, {}};
    const __bf16* vb = vT + ((size_t)(b * NHH + h)) * DHD * SS;
    for (int kc = 0; kc < SS / 32; ++kc) {
        v16bf pa  = load_a_frag_lds_f32(sS, SS, kc * 32, lane);
        v16bf vf0 = load_b_frag(vb, SS, 0,  kc * 32, lane);
        v16bf vf1 = load_b_frag(vb, SS, 16, kc * 32, lane);
        v16bf vf2 = load_b_frag(vb, SS, 32, kc * 32, lane);
        v16bf vf3 = load_b_frag(vb, SS, 48, kc * 32, lane);
        cacc[0] = wmma_bf16(pa, vf0, cacc[0]);
        cacc[1] = wmma_bf16(pa, vf1, cacc[1]);
        cacc[2] = wmma_bf16(pa, vf2, cacc[2]);
        cacc[3] = wmma_bf16(pa, vf3, cacc[3]);
    }
#pragma unroll
    for (int j = 0; j < 4; ++j) {
#pragma unroll
        for (int r = 0; r < 8; ++r) {
            int mrow  = r + 8 * g;
            float rsc = __shfl(inv, mrow);
            float val = cacc[j][r] * rsc;
            ctx[(size_t)(tok0 + mrow) * HH + h * DHD + j * 16 + nn] = (__bf16)val;
        }
    }
}

// ---------------------------------------------------------------------------
// Tail: a = h2 @ We[:H],  c = h2 @ We[H:]
// ---------------------------------------------------------------------------
__global__ void span_tail_kernel(const float* __restrict__ h2,
                                 const float* __restrict__ We,
                                 float* __restrict__ a, float* __restrict__ c) {
    int t = blockIdx.x;
    const float* row = h2 + (size_t)t * HH;
    float a0 = 0.f, a1 = 0.f, c0 = 0.f, c1 = 0.f;
    for (int j = threadIdx.x; j < HH; j += blockDim.x) {
        float x = row[j];
        a0 += x * We[j * 2 + 0];
        a1 += x * We[j * 2 + 1];
        c0 += x * We[(HH + j) * 2 + 0];
        c1 += x * We[(HH + j) * 2 + 1];
    }
    a0 = block_reduce_sum(a0);
    a1 = block_reduce_sum(a1);
    c0 = block_reduce_sum(c0);
    c1 = block_reduce_sum(c1);
    if (threadIdx.x == 0) {
        a[t * 2 + 0] = a0; a[t * 2 + 1] = a1;
        c[t * 2 + 0] = c0; c[t * 2 + 1] = c1;
    }
}

__global__ void span_end_kernel(const float* __restrict__ a,
                                const float* __restrict__ c,
                                const float* __restrict__ be,
                                float* __restrict__ out, long long total) {
    float be0 = be[0], be1 = be[1];
    for (long long i = blockIdx.x * (long long)blockDim.x + threadIdx.x;
         i < total; i += (long long)gridDim.x * blockDim.x) {
        int ch = (int)(i & 1);
        long long r = i >> 1;
        int j  = (int)(r & 1023);
        int qi = (int)((r >> 10) & 1023);
        int b  = (int)(r >> 20);
        out[i] = a[((b << 10) + qi) * 2 + ch] + c[((b << 10) + j) * 2 + ch] +
                 (ch ? be1 : be0);
    }
}

// ---------------------------------------------------------------------------
// Host side
// ---------------------------------------------------------------------------
extern "C" void kernel_launch(void* const* d_in, const int* in_sizes, int n_in,
                              void* d_out, int out_size, void* d_ws, size_t ws_size,
                              hipStream_t stream) {
    (void)in_sizes; (void)n_in; (void)ws_size;

    const int*   ids   = (const int*)  d_in[0];
    const int*   amask = (const int*)  d_in[1];
    const float* embed = (const float*)d_in[2];
    const float* ln0_g = (const float*)d_in[3];
    const float* ln0_b = (const float*)d_in[4];
    const float* Wq    = (const float*)d_in[5];
    const float* bq    = (const float*)d_in[6];
    const float* Wk    = (const float*)d_in[7];
    const float* bk    = (const float*)d_in[8];
    const float* Wv    = (const float*)d_in[9];
    const float* bv    = (const float*)d_in[10];
    const float* Wo    = (const float*)d_in[11];
    const float* bo    = (const float*)d_in[12];
    const float* ln1_g = (const float*)d_in[13];
    const float* ln1_b = (const float*)d_in[14];
    const float* Wf1   = (const float*)d_in[15];
    const float* bf1   = (const float*)d_in[16];
    const float* Wf2   = (const float*)d_in[17];
    const float* bf2   = (const float*)d_in[18];
    const float* ln2_g = (const float*)d_in[19];
    const float* ln2_b = (const float*)d_in[20];
    /* Ws (21), bs (22) unused: span_start_logits deleted in reference */
    const float* We    = (const float*)d_in[23];
    const float* be    = (const float*)d_in[24];

    char* w = (char*)d_ws;
    auto alloc = [&](size_t bytes) {
        char* p = w;
        w += (bytes + 255) & ~(size_t)255;
        return p;
    };
    float*  bufA  = (float*) alloc((size_t)MTOK * HH * 4);
    float*  bufB  = (float*) alloc((size_t)MTOK * HH * 4);
    __bf16* hb    = (__bf16*)alloc((size_t)MTOK * HH * 2);
    __bf16* qb    = (__bf16*)alloc((size_t)MTOK * HH * 2);
    __bf16* kbf   = (__bf16*)alloc((size_t)MTOK * HH * 2);
    __bf16* vT    = (__bf16*)alloc((size_t)MTOK * HH * 2);
    __bf16* ctxb  = (__bf16*)alloc((size_t)MTOK * HH * 2);
    __bf16* g1b   = (__bf16*)alloc((size_t)MTOK * DFFC * 2);
    __bf16* WqT   = (__bf16*)alloc((size_t)HH * HH * 2);
    __bf16* WkT   = (__bf16*)alloc((size_t)HH * HH * 2);
    __bf16* WvT   = (__bf16*)alloc((size_t)HH * HH * 2);
    __bf16* WoT   = (__bf16*)alloc((size_t)HH * HH * 2);
    __bf16* Wf1T  = (__bf16*)alloc((size_t)HH * DFFC * 2);
    __bf16* Wf2T  = (__bf16*)alloc((size_t)DFFC * HH * 2);
    float*  a_arr = (float*) alloc((size_t)MTOK * 2 * 4);
    float*  c_arr = (float*) alloc((size_t)MTOK * 2 * 4);

    wt_convert_kernel<<<1024, 256, 0, stream>>>(Wq,  WqT,  HH,   HH);
    wt_convert_kernel<<<1024, 256, 0, stream>>>(Wk,  WkT,  HH,   HH);
    wt_convert_kernel<<<1024, 256, 0, stream>>>(Wv,  WvT,  HH,   HH);
    wt_convert_kernel<<<1024, 256, 0, stream>>>(Wo,  WoT,  HH,   HH);
    wt_convert_kernel<<<2048, 256, 0, stream>>>(Wf1, Wf1T, HH,   DFFC);
    wt_convert_kernel<<<2048, 256, 0, stream>>>(Wf2, Wf2T, DFFC, HH);

    embed_ln_kernel<<<MTOK, 256, 0, stream>>>(ids, embed, ln0_g, ln0_b, bufA, hb);

    dim3 gH (HH   / 256, MTOK / 64);   // (3, 64)
    dim3 gF (DFFC / 256, MTOK / 64);   // (12, 64)

    gemm_wmma_kernel<<<gH, 256, 0, stream>>>(hb, WqT, bq, MTOK, HH, HH,
                                             nullptr, qb,  nullptr, nullptr, 0);
    gemm_wmma_kernel<<<gH, 256, 0, stream>>>(hb, WkT, bk, MTOK, HH, HH,
                                             nullptr, kbf, nullptr, nullptr, 0);
    gemm_wmma_kernel<<<gH, 256, 0, stream>>>(hb, WvT, bv, MTOK, HH, HH,
                                             nullptr, nullptr, nullptr, vT, 0);

    attention_kernel<<<dim3(SS / 16, NHH, BB), 32, 16 * SS * sizeof(float),
                       stream>>>(qb, kbf, vT, amask, ctxb);

    gemm_wmma_kernel<<<gH, 256, 0, stream>>>(ctxb, WoT, bo, MTOK, HH, HH,
                                             bufB, nullptr, bufA, nullptr, 0);
    ln_kernel<<<MTOK, 256, 0, stream>>>(bufB, ln1_g, ln1_b, bufA, hb);

    gemm_wmma_kernel<<<gF, 256, 0, stream>>>(hb, Wf1T, bf1, MTOK, DFFC, HH,
                                             nullptr, g1b, nullptr, nullptr, 1);
    gemm_wmma_kernel<<<gH, 256, 0, stream>>>(g1b, Wf2T, bf2, MTOK, HH, DFFC,
                                             bufB, nullptr, bufA, nullptr, 0);
    ln_kernel<<<MTOK, 256, 0, stream>>>(bufB, ln2_g, ln2_b, bufA, nullptr);

    span_tail_kernel<<<MTOK, 256, 0, stream>>>(bufA, We, a_arr, c_arr);
    span_end_kernel<<<8192, 256, 0, stream>>>(a_arr, c_arr, be, (float*)d_out,
                                              (long long)out_size);
}
